// LearnableCorrBlock_66451734003973
// MI455X (gfx1250) — matrix operand order repaired and checked by reading, not compiled, not run on hardware
//
#include <hip/hip_runtime.h>
#include <hip/hip_bf16.h>

#define DIMC 256
#define HDIM 64
#define WDIM 64
#define HW   4096            // 64*64
#define NB   2
#define RAD  4
#define NCH  324             // 4 levels * 81

typedef __bf16 v16bf __attribute__((ext_vector_type(16)));
typedef float  v8f   __attribute__((ext_vector_type(8)));
typedef unsigned short ushort8v __attribute__((ext_vector_type(8)));
typedef unsigned short ushort4v __attribute__((ext_vector_type(4)));
typedef float float4v __attribute__((ext_vector_type(4)));

union FragBF { unsigned short u[16]; ushort8v h[2]; v16bf v; };

__device__ __forceinline__ unsigned short f2bf(float f) {
    unsigned int u = __float_as_uint(f);
    u += 0x7FFFu + ((u >> 16) & 1u);          // round-to-nearest-even
    return (unsigned short)(u >> 16);
}

// ---------------- metric construction -----------------------------------

// Maug = [ (I - skew) | I ]  (256 x 512),  IpS = I + skew
__global__ void k_build_aug(const float* __restrict__ rawP,
                            float* __restrict__ Maug,
                            float* __restrict__ IpS) {
    int r = blockIdx.x, c = threadIdx.x;
    float urc = (c >= r) ? rawP[r * DIMC + c] : 0.f;   // triu(rawP)[r][c]
    float ucr = (r >= c) ? rawP[c * DIMC + r] : 0.f;   // triu(rawP)[c][r]
    float s   = 0.5f * (urc - ucr);
    float id  = (r == c) ? 1.f : 0.f;
    Maug[r * 2 * DIMC + c]        = id - s;
    Maug[r * 2 * DIMC + DIMC + c] = id;
    IpS [r * DIMC + c]            = id + s;
}

// Gauss-Jordan on the 256x512 augmented matrix (single workgroup)
__global__ void k_gj(float* __restrict__ M) {
    int t = threadIdx.x;
    for (int k = 0; k < DIMC; ++k) {
        __syncthreads();
        float piv = M[k * 2 * DIMC + k];
        float inv = 1.f / piv;
        __syncthreads();
        M[k * 2 * DIMC + t]        *= inv;
        M[k * 2 * DIMC + DIMC + t] *= inv;
        __syncthreads();
        if (t != k) {
            float f = M[t * 2 * DIMC + k];
            for (int c = 0; c < 2 * DIMC; ++c)
                M[t * 2 * DIMC + c] -= f * M[k * 2 * DIMC + c];
        }
    }
}

// P = (I + skew) @ inv(I - skew)   (inverse lives in Maug[:, 256:512])
__global__ void k_P(const float* __restrict__ IpS,
                    const float* __restrict__ Maug,
                    float* __restrict__ P) {
    int r = blockIdx.x, c = threadIdx.x;
    float acc = 0.f;
    for (int j = 0; j < DIMC; ++j)
        acc += IpS[r * DIMC + j] * Maug[j * 2 * DIMC + DIMC + c];
    P[r * DIMC + c] = acc;
}

__global__ void k_diag(const float* __restrict__ rawD, float* __restrict__ dvec) {
    int i = threadIdx.x;
    float t = atanf(rawD[i]) * 0.636619772367581343f; // 2/pi
    dvec[i] = (1.f + t) / (1.f - t);
}

// W[r][c] = sum_j P[j][r] * d[j] * P[j][c]
__global__ void k_W(const float* __restrict__ P,
                    const float* __restrict__ dvec,
                    float* __restrict__ Wm) {
    int r = blockIdx.x, c = threadIdx.x;
    float acc = 0.f;
    for (int j = 0; j < DIMC; ++j)
        acc += P[j * DIMC + r] * dvec[j] * P[j * DIMC + c];
    Wm[r * DIMC + c] = acc;
}

// ---------------- operand preparation -----------------------------------

// float4 -> bf16x4 (vectorized; count in float4 units)
__global__ void k_cvt_bf4(const float* __restrict__ src,
                          unsigned short* __restrict__ dst, int n4) {
    int i = blockIdx.x * blockDim.x + threadIdx.x;
    if (i >= n4) return;
    float4v v = *(const float4v*)(src + 4 * (size_t)i);
    ushort4v o;
    o.x = f2bf(v.x); o.y = f2bf(v.y); o.z = f2bf(v.z); o.w = f2bf(v.w);
    *(ushort4v*)(dst + 4 * (size_t)i) = o;
}

// G[b][c][p] = (1/16) * sum_k W[c][k] * f2[b][k][p], stored bf16
__global__ void k_G(const float* __restrict__ Wm,
                    const float* __restrict__ f2,
                    unsigned short* __restrict__ G) {
    int bc = blockIdx.y;               // b*256 + c
    int b  = bc >> 8, c = bc & 255;
    int p  = blockIdx.x * 256 + threadIdx.x;
    const float* Wrow = Wm + c * DIMC;
    const float* f2b  = f2 + (size_t)b * DIMC * HW + p;
    float acc = 0.f;
    for (int k = 0; k < DIMC; ++k)
        acc += Wrow[k] * f2b[(size_t)k * HW];
    G[(size_t)bc * HW + p] = f2bf(acc * 0.0625f);  // fold 1/sqrt(256)
}

// bf16 transpose per batch: out[b][p][c] = in[b][c][p]  (K becomes contiguous)
__global__ void k_tr(const unsigned short* __restrict__ in,   // [B][256][4096]
                     unsigned short* __restrict__ out) {      // [B][4096][256]
    __shared__ unsigned short tile[32][33];
    int b  = blockIdx.z;
    int c0 = blockIdx.y * 32;            // over DIMC
    int p0 = blockIdx.x * 32;            // over HW
    int tx = threadIdx.x & 31, ty = threadIdx.x >> 5;   // 32 x 8
    #pragma unroll
    for (int q = 0; q < 4; ++q)
        tile[ty + q * 8][tx] =
            in[((size_t)b * DIMC + (c0 + ty + q * 8)) * HW + p0 + tx];
    __syncthreads();
    #pragma unroll
    for (int q = 0; q < 4; ++q)
        out[((size_t)b * HW + (p0 + ty + q * 8)) * DIMC + c0 + tx] =
            tile[tx][ty + q * 8];
}

// ---------------- main WMMA GEMM: corr = f1^T @ G ------------------------
// Operands pre-transposed to [p][c] so K is contiguous. Block tile 64x128,
// 8 waves as 2x4 grid of 32x32 wave tiles, 2x2 v_wmma_f32_16x16x32_bf16 each.
// All staging is b128 loads/stores; fragment fills are contiguous K runs:
//   A lane: chunks at K = 8*lh and 16+8*lh;  B lane: chunks at K = 8*lh and 16+8*lh.

__global__ void __launch_bounds__(256)
k_corr(const unsigned short* __restrict__ At,    // f1^T bf16 [B][4096][256]
       const unsigned short* __restrict__ Bt,    // G^T  bf16 [B][4096][256]
       float* __restrict__ corr) {               // [B][4096][4096]
    __shared__ __align__(32) unsigned short Ast[64 * 32];    // [i][kk]
    __shared__ __align__(32) unsigned short Bst[128 * 32];   // [j][kk]

    const int b    = blockIdx.z;
    const int i0   = blockIdx.y * 64;
    const int j0   = blockIdx.x * 128;
    const int tid  = threadIdx.x;
    const int lane = tid & 31, wave = tid >> 5;
    const int m0   = (wave & 1) * 32;            // wave row in block tile
    const int n0   = (wave >> 1) * 32;           // wave col in block tile
    const int lm   = lane & 15, lh = lane >> 4;

    v8f acc[4] = {};

    const unsigned short* Abase = At + (size_t)b * HW * DIMC;
    const unsigned short* Bbase = Bt + (size_t)b * HW * DIMC;

    const int iA = tid >> 2;              // 0..63
    const int kA = (tid & 3) * 8;         // 0..24

    for (int k0 = 0; k0 < DIMC; k0 += 32) {
        // stage A (64 rows x 32 K): b128 load -> b128 store, no scatter
        *(ushort8v*)&Ast[iA * 32 + kA] =
            *(const ushort8v*)(Abase + (size_t)(i0 + iA) * DIMC + k0 + kA);
        // stage B (128 rows x 32 K): two b128 load/store per thread
        #pragma unroll
        for (int q = 0; q < 2; ++q) {
            int idx = q * 256 + tid;
            int j = idx >> 2, kb = (idx & 3) * 8;
            *(ushort8v*)&Bst[j * 32 + kb] =
                *(const ushort8v*)(Bbase + (size_t)(j0 + j) * DIMC + k0 + kb);
        }
        if (k0 + 32 < DIMC) {                    // prefetch next K chunk
            __builtin_prefetch(Abase + (size_t)(i0 + iA) * DIMC + k0 + 32, 0, 3);
            __builtin_prefetch(Bbase + (size_t)(j0 + iA) * DIMC + k0 + 32, 0, 3);
        }
        __syncthreads();

        FragBF a[2], bf[2];
        #pragma unroll
        for (int tI = 0; tI < 2; ++tI) {
            int row = m0 + tI * 16 + lm;
            a[tI].h[0] = *(const ushort8v*)&Ast[row * 32 + 8 * lh];
            a[tI].h[1] = *(const ushort8v*)&Ast[row * 32 + 8 * lh + 16];
        }
        #pragma unroll
        for (int tJ = 0; tJ < 2; ++tJ) {
            int col = n0 + tJ * 16 + lm;
            bf[tJ].h[0] = *(const ushort8v*)&Bst[col * 32 + 8 * lh];
            bf[tJ].h[1] = *(const ushort8v*)&Bst[col * 32 + 8 * lh + 16];
        }
        #pragma unroll
        for (int tI = 0; tI < 2; ++tI)
            #pragma unroll
            for (int tJ = 0; tJ < 2; ++tJ)
                acc[tI * 2 + tJ] = __builtin_amdgcn_wmma_f32_16x16x32_bf16(
                    false, a[tI].v, false, bf[tJ].v,
                    (short)0, acc[tI * 2 + tJ], false, false);
        __syncthreads();
    }

    // store: C/D layout — VGPR r: lanes0-15 M=r, lanes16-31 M=8+r; N=lm
    float* cb = corr + (size_t)b * HW * HW;
    #pragma unroll
    for (int tI = 0; tI < 2; ++tI)
        #pragma unroll
        for (int tJ = 0; tJ < 2; ++tJ) {
            int col = j0 + n0 + tJ * 16 + lm;
            #pragma unroll
            for (int r = 0; r < 8; ++r) {
                int row = i0 + m0 + tI * 16 + lh * 8 + r;
                cb[(size_t)row * HW + col] = acc[tI * 2 + tJ][r];
            }
        }
}

// ---------------- pyramid + sampling -------------------------------------

__global__ void k_pool(const float* __restrict__ in, float* __restrict__ out,
                       int hin, int win, long total) {
    size_t idx = (size_t)blockIdx.x * blockDim.x + threadIdx.x;
    if (idx >= (size_t)total) return;
    int wo = win >> 1, ho = hin >> 1;
    int x = (int)(idx % wo);
    size_t t = idx / wo;
    int y = (int)(t % ho);
    size_t n = t / ho;
    const float* p = in + n * (size_t)hin * win + (size_t)(2 * y) * win + 2 * x;
    out[idx] = 0.25f * (p[0] + p[1] + p[win] + p[win + 1]);
}

__device__ __forceinline__ float fetch0(const float* img, int xi, int yi,
                                        int wl, int hl) {
    if (xi < 0 || xi >= wl || yi < 0 || yi >= hl) return 0.f;
    return img[yi * wl + xi];
}

__global__ void k_sample(const float* __restrict__ coords,
                         const float* __restrict__ p0, const float* __restrict__ p1,
                         const float* __restrict__ p2, const float* __restrict__ p3,
                         float* __restrict__ out, long total) {
    size_t idx = (size_t)blockIdx.x * blockDim.x + threadIdx.x;
    if (idx >= (size_t)total) return;
    int w = (int)(idx & 63);
    size_t t = idx >> 6;
    int h = (int)(t & 63);
    t >>= 6;
    int ch = (int)(t % NCH);
    int b  = (int)(t / NCH);
    int level = ch / 81, rem = ch % 81;
    int ax = rem / 9, by = rem % 9;                 // x <- dgy axis, y <- dgx axis
    float cx = coords[((size_t)(b * 2 + 0) * HDIM + h) * WDIM + w];
    float cy = coords[((size_t)(b * 2 + 1) * HDIM + h) * WDIM + w];
    float sc = 1.0f / (float)(1 << level);
    float x = cx * sc + (float)(ax - RAD);
    float y = cy * sc + (float)(by - RAD);
    int hl = HDIM >> level, wl = WDIM >> level;
    const float* lv = (level == 0) ? p0 : (level == 1) ? p1 : (level == 2) ? p2 : p3;
    const float* img = lv + (size_t)(b * HW + h * WDIM + w) * hl * wl;
    float x0f = floorf(x), y0f = floorf(y);
    int x0 = (int)x0f, y0 = (int)y0f;
    float wx1 = x - x0f, wx0 = 1.f - wx1;
    float wy1 = y - y0f, wy0 = 1.f - wy1;
    float v = wy0 * wx0 * fetch0(img, x0,     y0,     wl, hl)
            + wy0 * wx1 * fetch0(img, x0 + 1, y0,     wl, hl)
            + wy1 * wx0 * fetch0(img, x0,     y0 + 1, wl, hl)
            + wy1 * wx1 * fetch0(img, x0 + 1, y0 + 1, wl, hl);
    out[idx] = v;
}

// ---------------- launch ---------------------------------------------------

extern "C" void kernel_launch(void* const* d_in, const int* in_sizes, int n_in,
                              void* d_out, int out_size, void* d_ws, size_t ws_size,
                              hipStream_t stream) {
    const float* fmap1  = (const float*)d_in[0];
    const float* fmap2  = (const float*)d_in[1];
    const float* coords = (const float*)d_in[2];
    const float* rawP   = (const float*)d_in[3];
    const float* rawD   = (const float*)d_in[4];
    float* out = (float*)d_out;

    char* ws = (char*)d_ws;
    size_t off = 0;
    auto carve = [&](size_t bytes) -> char* {
        char* p = ws + off;
        off = (off + bytes + 255) & ~(size_t)255;
        return p;
    };
    float*          Maug = (float*)carve((size_t)DIMC * 2 * DIMC * 4);
    float*          IpS  = (float*)carve((size_t)DIMC * DIMC * 4);
    float*          Pm   = (float*)carve((size_t)DIMC * DIMC * 4);
    float*          dvec = (float*)carve(DIMC * 4);
    float*          Wm   = (float*)carve((size_t)DIMC * DIMC * 4);
    unsigned short* f1bf = (unsigned short*)carve((size_t)NB * DIMC * HW * 2);
    unsigned short* Gbf  = (unsigned short*)carve((size_t)NB * DIMC * HW * 2);
    unsigned short* f1t  = (unsigned short*)carve((size_t)NB * DIMC * HW * 2);
    unsigned short* Gt   = (unsigned short*)carve((size_t)NB * DIMC * HW * 2);
    float*          corr = (float*)carve((size_t)NB * HW * HW * 4);
    float*          pyr1 = (float*)carve((size_t)NB * HW * 32 * 32 * 4);
    float*          pyr2 = (float*)carve((size_t)NB * HW * 16 * 16 * 4);
    float*          pyr3 = (float*)carve((size_t)NB * HW * 8 * 8 * 4);

    // metric
    k_build_aug<<<DIMC, DIMC, 0, stream>>>(rawP, Maug, IpS);
    k_gj<<<1, DIMC, 0, stream>>>(Maug);
    k_P<<<DIMC, DIMC, 0, stream>>>(IpS, Maug, Pm);
    k_diag<<<1, DIMC, 0, stream>>>(rawD, dvec);
    k_W<<<DIMC, DIMC, 0, stream>>>(Pm, dvec, Wm);

    // operands (convert, apply metric, transpose so K is contiguous)
    int nf1 = NB * DIMC * HW;
    k_cvt_bf4<<<(nf1 / 4 + 255) / 256, 256, 0, stream>>>(fmap1, f1bf, nf1 / 4);
    k_G<<<dim3(HW / 256, NB * DIMC), 256, 0, stream>>>(Wm, fmap2, Gbf);
    k_tr<<<dim3(HW / 32, DIMC / 32, NB), 256, 0, stream>>>(f1bf, f1t);
    k_tr<<<dim3(HW / 32, DIMC / 32, NB), 256, 0, stream>>>(Gbf, Gt);

    // correlation GEMM (WMMA bf16 -> f32)
    k_corr<<<dim3(HW / 128, HW / 64, NB), 256, 0, stream>>>(f1t, Gt, corr);

    // pyramid
    long t1 = (long)NB * HW * 32 * 32;
    long t2 = (long)NB * HW * 16 * 16;
    long t3 = (long)NB * HW * 8 * 8;
    k_pool<<<(unsigned)((t1 + 255) / 256), 256, 0, stream>>>(corr, pyr1, 64, 64, t1);
    k_pool<<<(unsigned)((t2 + 255) / 256), 256, 0, stream>>>(pyr1, pyr2, 32, 32, t2);
    k_pool<<<(unsigned)((t3 + 255) / 256), 256, 0, stream>>>(pyr2, pyr3, 16, 16, t3);

    // multi-level window lookup
    long ts = (long)NB * NCH * HW;
    k_sample<<<(unsigned)((ts + 255) / 256), 256, 0, stream>>>(coords, corr, pyr1,
                                                               pyr2, pyr3, out, ts);
}
